// ConvexUpSample_83184926589555
// MI455X (gfx1250) — matrix operand order repaired and checked by reading, not compiled, not run on hardware
//
#include <hip/hip_runtime.h>
#include <hip/hip_bf16.h>

typedef __attribute__((ext_vector_type(16))) _Float16 v16h;
typedef __attribute__((ext_vector_type(8)))  float    v8f;
typedef __attribute__((ext_vector_type(4)))  int      i32x4;

// CDNA5 async global->LDS staging (ASYNCcnt-tracked), if this toolchain has it.
#if __has_builtin(__builtin_amdgcn_global_load_async_to_lds_b128) && \
    __has_builtin(__builtin_amdgcn_global_load_async_to_lds_b32) &&  \
    __has_builtin(__builtin_amdgcn_s_wait_asynccnt)
#define HAS_ASYNC_LDS 1
typedef __attribute__((address_space(1))) i32x4 g_i32x4;  // global b128 payload
typedef __attribute__((address_space(3))) i32x4 s_i32x4;  // LDS b128 payload
typedef __attribute__((address_space(1))) int   g_i32;    // global b32 payload
typedef __attribute__((address_space(3))) int   s_i32;    // LDS b32 payload
#else
#define HAS_ASYNC_LDS 0
#endif

constexpr int BB = 8, HH = 56, WW = 56, HW = HH * WW;
constexpr int PIX = BB * HW;          // 25088
constexpr int HO = HH * 4, WO = WW * 4;
constexpr int CIN0 = 128, CMID = 256, COUT3 = 16, CMASK = 144;

// A-row K-segment permutation: store 8-half segments in order [0,16,8,24]
// so each lane's 16-half WMMA A-fragment is one contiguous 32-byte LDS read.
// (swaps bits 3<->4 of the half-granular column offset; cols are 0/8/16/24)
__device__ __forceinline__ int permACol(int c) {
  return ((c & 8) << 1) | ((c & 16) >> 1);
}

// ---------------------------------------------------------------------------
// Layout conversion: NCHW f32 -> NHWC f16
// ---------------------------------------------------------------------------
__global__ __launch_bounds__(256) void nchw_to_nhwc_f16(const float* __restrict__ x,
                                                        _Float16* __restrict__ xh) {
  int idx = blockIdx.x * 256 + threadIdx.x;            // over PIX*128, NHWC order
  if (idx >= PIX * CIN0) return;
  int c = idx % CIN0;
  int p = idx / CIN0;
  int b = p / HW;
  int hw = p % HW;
  xh[idx] = (_Float16)x[((size_t)b * CIN0 + c) * HW + hw];
}

// ---------------------------------------------------------------------------
// Weight transpose+convert: (tap, cin, cout) f32 -> (tap, cout, cin) f16
// ---------------------------------------------------------------------------
__global__ __launch_bounds__(256) void wT_f16(const float* __restrict__ src,
                                              _Float16* __restrict__ dst,
                                              int taps, int cin, int cout) {
  int idx = blockIdx.x * 256 + threadIdx.x;            // dst-order index
  int total = taps * cin * cout;
  if (idx >= total) return;
  int ci = idx % cin;
  int rest = idx / cin;
  int co = rest % cout;
  int t  = rest / cout;
  dst[idx] = (_Float16)src[((size_t)t * cin + ci) * cout + co];
}

// ---------------------------------------------------------------------------
// Implicit-GEMM conv (SAME, 3x3 or 1x1) with v_wmma_f32_16x16x32_f16.
//   in  : NHWC f16 activations [PIX][CIN]
//   wT  : f16 weights [NTAPS][COUT][CIN]
//   out : [PIX][COUT], f16 (OUT_F16) or f32
//
// 256 threads = 8 waves arranged WAVES_M x WAVES_N; each wave owns a
// (WM*16) x (WN*16) output tile. K-chunks (32-deep) are software-pipelined
// through double-buffered LDS. Staging uses CDNA5 async global->LDS DMA when
// available; A rows are stored K-segment-permuted so each A fragment is a
// single contiguous 32B LDS read.
// ---------------------------------------------------------------------------
template<int CIN, int COUT, int NTAPS, int WAVES_M, int WAVES_N, int WM, int WN,
         bool RELU, bool OUT_F16>
__global__ __launch_bounds__(256) void conv_wmma(const _Float16* __restrict__ in,
                                                 const _Float16* __restrict__ wT,
                                                 const float* __restrict__ bias,
                                                 void* __restrict__ outv) {
  constexpr int BM = WAVES_M * WM * 16;
  constexpr int BN = WAVES_N * WN * 16;
  constexpr int PITCH = 48;              // halves per LDS row: 96 B, keeps 16/32B alignment
  constexpr int KCH = CIN / 32;          // K-chunks per tap
  constexpr int NCHUNK = NTAPS * KCH;
  static_assert(WAVES_M * WAVES_N == 8, "8 waves per block");

  __shared__ alignas(32) _Float16 As[2][BM * PITCH];
  __shared__ alignas(32) _Float16 Bs[2][BN * PITCH];

  const int tid    = threadIdx.x;
  const int wave   = tid >> 5;
  const int lane   = tid & 31;
  const int lhalf  = lane & 15;
  const int hiHalf = lane >> 4;          // 0 or 1

  const int m0 = blockIdx.x * BM;        // pixel base
  const int n0 = blockIdx.y * BN;        // out-channel base
  const int waveM = wave / WAVES_N;
  const int waveN = wave % WAVES_N;

  const v8f vzero = {0.f, 0.f, 0.f, 0.f, 0.f, 0.f, 0.f, 0.f};
  v8f acc[WM][WN];
#pragma unroll
  for (int i = 0; i < WM; ++i)
#pragma unroll
    for (int j = 0; j < WN; ++j) acc[i][j] = vzero;

  // ---- staging geometry ----
  constexpr int A_HPT = (BM * 32) / 256;     // halves/thread: 8 (BM=64) or 32 (BM=256)
  constexpr int A_NV  = A_HPT / 8;           // b128 transfers per thread
  static_assert(A_HPT % 8 == 0, "A staging must be b128-granular");
  constexpr int A_TPR = 32 / ((A_HPT < 32) ? A_HPT : 32);
  const int aRow = tid / A_TPR;
  const int aCol = (tid % A_TPR) * ((A_HPT < 32) ? A_HPT : 32);
  const int ap  = m0 + aRow;
  const int ab  = ap / HW;
  const int ahw = ap % HW;
  const int ah  = ahw / WW;
  const int aw  = ahw % WW;

  constexpr int B_HPT = (BN * 32) / 256;     // 16 (BN=128) or 2 (BN=16)
  constexpr int B_NV  = (B_HPT >= 8) ? (B_HPT / 8) : 0;
  constexpr int B_TPR = 32 / B_HPT;
  const int bRow = tid / B_TPR;
  const int bCol = (tid % B_TPR) * B_HPT;

#if !HAS_ASYNC_LDS
  uint4 aReg[A_NV];
  uint4 bReg[(B_NV > 0) ? B_NV : 1];
  unsigned bRegS = 0;
#endif

  // Stage A tile (BM x 32 halves, K-segments permuted) of `chunk` into buffer `buf`.
  auto stageA = [&](int chunk, int buf) {
    const int tap = chunk / KCH;
    const int kc  = (chunk - tap * KCH) * 32;
    const int dy = (NTAPS == 9) ? (tap / 3 - 1) : 0;
    const int dx = (NTAPS == 9) ? (tap % 3 - 1) : 0;
    const int iy = ah + dy, ix = aw + dx;
    const bool valid = (iy >= 0) && (iy < HH) && (ix >= 0) && (ix < WW);
    const _Float16* s = in + ((size_t)((ab * HH + iy) * WW + ix)) * CIN + kc + aCol;
    _Float16* drow = &As[buf][aRow * PITCH];
#if HAS_ASYNC_LDS
    if (valid) {
#pragma unroll
      for (int q = 0; q < A_NV; ++q)
        __builtin_amdgcn_global_load_async_to_lds_b128(
            (g_i32x4*)(s + q * 8), (s_i32x4*)(drow + permACol(aCol + q * 8)), 0, 0);
    } else {
      const uint4 z = make_uint4(0u, 0u, 0u, 0u);
#pragma unroll
      for (int q = 0; q < A_NV; ++q)
        *(uint4*)(drow + permACol(aCol + q * 8)) = z;
    }
#else
    const uint4 z = make_uint4(0u, 0u, 0u, 0u);
#pragma unroll
    for (int q = 0; q < A_NV; ++q)
      aReg[q] = valid ? *(const uint4*)(s + q * 8) : z;
#pragma unroll
    for (int q = 0; q < A_NV; ++q)
      *(uint4*)(drow + permACol(aCol + q * 8)) = aReg[q];
#endif
  };

  // Stage B tile (BN x 32 halves, linear K) of `chunk` into buffer `buf`.
  auto stageB = [&](int chunk, int buf) {
    const int tap = chunk / KCH;
    const int kc  = (chunk - tap * KCH) * 32;
    const _Float16* s = wT + ((size_t)tap * COUT + n0 + bRow) * CIN + kc + bCol;
    _Float16* d = &Bs[buf][bRow * PITCH + bCol];
#if HAS_ASYNC_LDS
    if constexpr (B_NV > 0) {
#pragma unroll
      for (int q = 0; q < B_NV; ++q)
        __builtin_amdgcn_global_load_async_to_lds_b128(
            (g_i32x4*)(s + q * 8), (s_i32x4*)(d + q * 8), 0, 0);
    } else {
      __builtin_amdgcn_global_load_async_to_lds_b32((g_i32*)s, (s_i32*)d, 0, 0);
    }
#else
    if constexpr (B_NV > 0) {
#pragma unroll
      for (int q = 0; q < B_NV; ++q)
        bReg[q] = *(const uint4*)(s + q * 8);
#pragma unroll
      for (int q = 0; q < B_NV; ++q)
        *(uint4*)(d + q * 8) = bReg[q];
    } else {
      bRegS = *(const unsigned*)s;
      *(unsigned*)d = bRegS;
    }
#endif
  };

  // ---- prologue: stage chunk 0 into buffer 0 ----
  stageA(0, 0);
  stageB(0, 0);
#if HAS_ASYNC_LDS
  __builtin_amdgcn_s_wait_asynccnt(0);
#endif
  __syncthreads();

  // ---- pipelined main loop: one barrier per K-chunk ----
#pragma unroll 2
  for (int chunk = 0; chunk < NCHUNK; ++chunk) {
    const int cur = chunk & 1;
    const bool hasNext = (chunk + 1) < NCHUNK;   // block-uniform
    if (hasNext) { stageA(chunk + 1, cur ^ 1); stageB(chunk + 1, cur ^ 1); }

    // A fragments: one contiguous 32B LDS read per fragment (pre-swizzled rows):
    // lane<16 -> K {0..7,16..23}; lane>=16 -> {8..15,24..31}
    v16h af[WM];
#pragma unroll
    for (int i = 0; i < WM; ++i) {
      const int arow = (waveM * WM + i) * 16 + lhalf;
      af[i] = *(const v16h*)&As[cur][arow * PITCH + hiHalf * 16];
    }
    // B fragments + WMMAs: lane<16 -> K {0..7,16..23} ; lane>=16 -> K {8..15,24..31}
#pragma unroll
    for (int j = 0; j < WN; ++j) {
      const int brow = (waveN * WN + j) * 16 + lhalf;
      const int bk0  = hiHalf * 8;
      v16h bfrag;
      ((uint4*)&bfrag)[0] = *(const uint4*)&Bs[cur][brow * PITCH + bk0];
      ((uint4*)&bfrag)[1] = *(const uint4*)&Bs[cur][brow * PITCH + 16 + bk0];
#pragma unroll
      for (int i = 0; i < WM; ++i)
        acc[i][j] = __builtin_amdgcn_wmma_f32_16x16x32_f16(
            false, af[i], false, bfrag, (short)0, acc[i][j], false, false);
    }

#if HAS_ASYNC_LDS
    __builtin_amdgcn_s_wait_asynccnt(0);   // our async writes to buf cur^1 landed
#endif
    __syncthreads();
  }

  // ---- epilogue: C/D layout = lane (N), vgpr v (M = v + 8*hiHalf) ----
#pragma unroll
  for (int j = 0; j < WN; ++j) {
    const int n = n0 + (waveN * WN + j) * 16 + lhalf;
    const float bv = bias[n];
#pragma unroll
    for (int i = 0; i < WM; ++i) {
      const int mb = m0 + (waveM * WM + i) * 16 + hiHalf * 8;
#pragma unroll
      for (int v = 0; v < 8; ++v) {
        float r = acc[i][j][v] + bv;
        if constexpr (RELU) r = fmaxf(r, 0.f);
        if constexpr (OUT_F16)
          ((_Float16*)outv)[(size_t)(mb + v) * COUT + n] = (_Float16)r;
        else
          ((float*)outv)[(size_t)(mb + v) * COUT + n] = r;
      }
    }
  }
}

// ---------------------------------------------------------------------------
// Softmax over 9 taps + convex combination + 4x upsample.
// Thread = (pixel, subpixel uv); out3 [PIX][16] f32, mask [PIX][144] f32.
// ---------------------------------------------------------------------------
__global__ __launch_bounds__(256) void combine_k(const float* __restrict__ out3,
                                                 const float* __restrict__ mask,
                                                 float* __restrict__ y) {
  int g = blockIdx.x * 256 + threadIdx.x;
  if (g >= PIX * 16) return;
  int uv = g & 15;
  int p  = g >> 4;
  int b  = p / HW;
  int hw = p % HW;
  int h  = hw / WW;
  int w  = hw % WW;

  // masked softmax over the 9 taps for this subpixel
  float mv[9];
  float mmax = -3.0e38f;
#pragma unroll
  for (int k = 0; k < 9; ++k) {
    mv[k] = 0.1f * mask[(size_t)p * CMASK + k * 16 + uv];
    mmax = fmaxf(mmax, mv[k]);
  }
  float sum = 0.f;
#pragma unroll
  for (int k = 0; k < 9; ++k) { mv[k] = __expf(mv[k] - mmax); sum += mv[k]; }
  const float inv = 1.f / sum;

  float acc[16];
#pragma unroll
  for (int oc = 0; oc < 16; ++oc) acc[oc] = 0.f;

#pragma unroll
  for (int k = 0; k < 9; ++k) {
    const int dy = k / 3 - 1, dx = k % 3 - 1;
    const int iy = h + dy, ix = w + dx;
    if (iy >= 0 && iy < HH && ix >= 0 && ix < WW) {
      const float4* src =
          (const float4*)(out3 + ((size_t)((b * HH + iy) * WW + ix)) * 16);
      const float wk = mv[k] * inv;
#pragma unroll
      for (int q = 0; q < 4; ++q) {
        float4 t = src[q];
        acc[q * 4 + 0] += wk * t.x;
        acc[q * 4 + 1] += wk * t.y;
        acc[q * 4 + 2] += wk * t.z;
        acc[q * 4 + 3] += wk * t.w;
      }
    }
  }

  const int u = uv >> 2, v = uv & 3;
#pragma unroll
  for (int oc = 0; oc < 16; ++oc)
    y[(((size_t)b * 16 + oc) * HO + (h * 4 + u)) * WO + (w * 4 + v)] = acc[oc];
}

// ---------------------------------------------------------------------------
extern "C" void kernel_launch(void* const* d_in, const int* in_sizes, int n_in,
                              void* d_out, int out_size, void* d_ws, size_t ws_size,
                              hipStream_t stream) {
  const float* x   = (const float*)d_in[0];
  const float* w1  = (const float*)d_in[1];
  const float* b1  = (const float*)d_in[2];
  const float* w2  = (const float*)d_in[3];
  const float* b2  = (const float*)d_in[4];
  const float* w3  = (const float*)d_in[5];
  const float* b3  = (const float*)d_in[6];
  const float* mw1 = (const float*)d_in[7];
  const float* mb1 = (const float*)d_in[8];
  const float* mw2 = (const float*)d_in[9];
  const float* mb2 = (const float*)d_in[10];
  float* y = (float*)d_out;

  // carve workspace
  char* ws = (char*)d_ws;
  size_t off = 0;
  auto carve = [&](size_t bytes) {
    void* p = ws + off;
    off = (off + bytes + 255) & ~(size_t)255;
    return p;
  };
  _Float16* xh    = (_Float16*)carve((size_t)PIX * CIN0 * 2);
  _Float16* act1  = (_Float16*)carve((size_t)PIX * CMID * 2);
  _Float16* act2  = (_Float16*)carve((size_t)PIX * CMID * 2);
  _Float16* mact1 = (_Float16*)carve((size_t)PIX * CMID * 2);
  float*    out3  = (float*)   carve((size_t)PIX * COUT3 * 4);
  float*    maskb = (float*)   carve((size_t)PIX * CMASK * 4);
  _Float16* wT1   = (_Float16*)carve((size_t)9 * CMID * CIN0 * 2);
  _Float16* wT2   = (_Float16*)carve((size_t)9 * CMID * CMID * 2);
  _Float16* wT3   = (_Float16*)carve((size_t)9 * COUT3 * CMID * 2);
  _Float16* mwT1  = (_Float16*)carve((size_t)9 * CMID * CIN0 * 2);
  _Float16* mwT2  = (_Float16*)carve((size_t)CMASK * CMID * 2);

  // 1) layout conversions
  nchw_to_nhwc_f16<<<(PIX * CIN0 + 255) / 256, 256, 0, stream>>>(x, xh);
  wT_f16<<<(9 * CIN0 * CMID + 255) / 256, 256, 0, stream>>>(w1,  wT1,  9, CIN0, CMID);
  wT_f16<<<(9 * CMID * CMID + 255) / 256, 256, 0, stream>>>(w2,  wT2,  9, CMID, CMID);
  wT_f16<<<(9 * CMID * COUT3 + 255) / 256, 256, 0, stream>>>(w3,  wT3,  9, CMID, COUT3);
  wT_f16<<<(9 * CIN0 * CMID + 255) / 256, 256, 0, stream>>>(mw1, mwT1, 9, CIN0, CMID);
  wT_f16<<<(CMID * CMASK + 255) / 256, 256, 0, stream>>>(mw2, mwT2, 1, CMID, CMASK);

  // 2) WMMA implicit-GEMM convolutions
  // big config: 2x4 waves, 32x32 per wave -> BM=64, BN=128
  conv_wmma<CIN0, CMID, 9, 2, 4, 2, 2, true, true>
      <<<dim3(PIX / 64, CMID / 128), 256, 0, stream>>>(xh, wT1, b1, act1);
  conv_wmma<CIN0, CMID, 9, 2, 4, 2, 2, true, true>
      <<<dim3(PIX / 64, CMID / 128), 256, 0, stream>>>(xh, mwT1, mb1, mact1);
  conv_wmma<CMID, CMID, 9, 2, 4, 2, 2, true, true>
      <<<dim3(PIX / 64, CMID / 128), 256, 0, stream>>>(act1, wT2, b2, act2);
  // small config: 8x1 waves, 32x16 per wave -> BM=256, BN=16
  conv_wmma<CMID, COUT3, 9, 8, 1, 2, 1, false, false>
      <<<dim3(PIX / 256, 1), 256, 0, stream>>>(act2, wT3, b3, out3);
  conv_wmma<CMID, CMASK, 1, 8, 1, 2, 1, false, false>
      <<<dim3(PIX / 256, CMASK / 16), 256, 0, stream>>>(mact1, mwT2, mb2, maskb);

  // 3) softmax + convex upsample
  combine_k<<<(PIX * 16 + 255) / 256, 256, 0, stream>>>(out3, maskb, y);
}
